// GCN_26989574488583
// MI455X (gfx1250) — matrix operand order repaired and checked by reading, not compiled, not run on hardware
//
#include <hip/hip_runtime.h>
#include <hip/hip_bf16.h>

// ---------------------------------------------------------------------------
// GENConv GNN (3 layers) + global mean pool + 2-layer head, for gfx1250.
// Dense matmuls use V_WMMA_F32_16X16X32_F16 (wave32 WMMA, f32 accumulate),
// with weight stripes staged f32->f16 into LDS in fragment-swizzled order.
// Segment-softmax aggregation uses exact uint-bit atomicMax (messages > 0)
// and fp32 L2 atomics for the two segment sums.
// ---------------------------------------------------------------------------

#define NN 50000      // nodes
#define EE 800000     // edges
#define IN_DIM 128
#define EDGE_DIM 32
#define OUT_DIM 10
#define NGRAPH 64

typedef __attribute__((ext_vector_type(16))) _Float16 v16h;
typedef __attribute__((ext_vector_type(8)))  float    v8f;

// ---------------------------------------------------------------------------
// WMMA GEMM: C[M,N] = act(A[M,K] @ W[K,N] + bias). Row-major f32 in/out.
// Block = 128 threads (4 waves); block tile = 64(M) x 64(N).
// Each wave owns one 16-row stripe and computes four 16x16 WMMA tiles per
// k-chunk, reusing a single A fragment. The 64-column B stripe is staged
// once per block into LDS as f16, pre-swizzled into per-lane fragment order
// so each fragment read is one 32-byte LDS vector load.
// Constraints (all satisfied here): K % 32 == 0, K <= 256, N % 64 == 0,
// M % 16 == 0.
//
// Fragment layouts used here (contiguous k, not AMD's table order; A and B match):
//  A: lane<16 holds row m=lane, k = kk+0..15; lane>=16: kk+16..31
//  B: lane holds column n=lane&15; lanes<16: k=kk+0..15, lanes>=16: kk+16..31
//  C/D: lane&15 = n, vgpr v -> m = 8*(lane>>4) + v
// ---------------------------------------------------------------------------
template <bool RELU>
__global__ void wmma_gemm_kernel(const float* __restrict__ A,
                                 const float* __restrict__ W,
                                 const float* __restrict__ bias,
                                 float* __restrict__ C,
                                 int M, int N, int K) {
  // LDS: per (k-chunk c, n-tile nt, lane l): 16 contiguous f16 = one B fragment
  // slice. Max K=256 -> 8 chunks * 4 ntiles * 32 lanes * 16 halfs = 32 KB.
  __shared__ __align__(32) _Float16 bs[8 * 4 * 32 * 16];

  const int lane = threadIdx.x & 31;
  const int wave = threadIdx.x >> 5;
  const int half = lane >> 4;   // 0 | 1
  const int l16  = lane & 15;
  const int col0 = blockIdx.y * 64;
  const int row0 = blockIdx.x * 64 + wave * 16;
  const int nchunks = K >> 5;

  // ---- cooperative B staging (uniform across block) ----
  for (int idx = threadIdx.x; idx < nchunks * 128; idx += blockDim.x) {
    const int c  = idx >> 7;          // k-chunk
    const int r  = idx & 127;         // nt*32 + lane
    const int nt = r >> 5;
    const int l  = r & 31;
    const int kbase = (c << 5) + ((l >> 4) << 4);
    const int n = col0 + (nt << 4) + (l & 15);
    _Float16* dstp = bs + (size_t)idx * 16;
    const float* wp = W + (size_t)kbase * N + n;
#pragma unroll
    for (int i = 0; i < 16; ++i) dstp[i] = (_Float16)wp[(size_t)i * N];
  }
  __syncthreads();

  if (row0 >= M) return;              // whole-wave guard (M % 16 == 0)

  v8f acc0 = {}, acc1 = {}, acc2 = {}, acc3 = {};
  const float* arowb = A + (size_t)(row0 + l16) * K + half * 16;
  for (int c = 0; c < nchunks; ++c) {
    const float* arow = arowb + (c << 5);
    __builtin_prefetch(arow + 32, 0, 0);          // next k-chunk
    const float4 a0 = *(const float4*)(arow);
    const float4 a1 = *(const float4*)(arow + 4);
    const float4 a2 = *(const float4*)(arow + 8);
    const float4 a3 = *(const float4*)(arow + 12);
    v16h a;
    a[0]  = (_Float16)a0.x; a[1]  = (_Float16)a0.y;
    a[2]  = (_Float16)a0.z; a[3]  = (_Float16)a0.w;
    a[4]  = (_Float16)a1.x; a[5]  = (_Float16)a1.y;
    a[6]  = (_Float16)a1.z; a[7]  = (_Float16)a1.w;
    a[8]  = (_Float16)a2.x; a[9]  = (_Float16)a2.y;
    a[10] = (_Float16)a2.z; a[11] = (_Float16)a2.w;
    a[12] = (_Float16)a3.x; a[13] = (_Float16)a3.y;
    a[14] = (_Float16)a3.z; a[15] = (_Float16)a3.w;

    const _Float16* bp = bs + ((size_t)(c << 7) + lane) * 16;
    const v16h b0 = *(const v16h*)(bp);
    const v16h b1 = *(const v16h*)(bp + 32 * 16);
    const v16h b2 = *(const v16h*)(bp + 64 * 16);
    const v16h b3 = *(const v16h*)(bp + 96 * 16);

    acc0 = __builtin_amdgcn_wmma_f32_16x16x32_f16(false, a, false, b0, (short)0, acc0, false, false);
    acc1 = __builtin_amdgcn_wmma_f32_16x16x32_f16(false, a, false, b1, (short)0, acc1, false, false);
    acc2 = __builtin_amdgcn_wmma_f32_16x16x32_f16(false, a, false, b2, (short)0, acc2, false, false);
    acc3 = __builtin_amdgcn_wmma_f32_16x16x32_f16(false, a, false, b3, (short)0, acc3, false, false);
  }

  // ---- epilogue: bias (+ optional relu), 4 n-tiles x 8 rows per lane ----
  const int mrow = row0 + half * 8;
  v8f accs[4] = {acc0, acc1, acc2, acc3};
#pragma unroll
  for (int nt = 0; nt < 4; ++nt) {
    const int n = col0 + (nt << 4) + l16;
    const float bn = bias ? bias[n] : 0.0f;
#pragma unroll
    for (int v = 0; v < 8; ++v) {
      float r = accs[nt][v] + bn;
      if (RELU) r = fmaxf(r, 0.0f);
      C[(size_t)(mrow + v) * N + n] = r;
    }
  }
}

// ---------------------------------------------------------------------------
// Pass A over edges: msg = relu(Hsrc[src] + Elin) + 1e-7 (stored in place over
// Elin), and exact segment-max via uint atomicMax (msg > 0, accum init = 0.0f,
// which also reproduces the reference's isinf(m)->0 substitution).
// One thread per (edge, 4 features).
// ---------------------------------------------------------------------------
__global__ void edge_msg_max_kernel(const float* __restrict__ Hsrc,
                                    float* __restrict__ Elin,
                                    const int* __restrict__ srcI,
                                    const int* __restrict__ dstI,
                                    unsigned int* __restrict__ mbits,
                                    int C, long long total) {
  long long tid = (long long)blockIdx.x * blockDim.x + threadIdx.x;
  if (tid >= total) return;
  const int c4 = C >> 2;
  const long long e = tid / c4;
  const int f = (int)(tid % c4) << 2;
  const int s = srcI[e];
  const int d = dstI[e];

  const float4 hv = *(const float4*)(Hsrc + (size_t)s * C + f);
  float4* ep = (float4*)(Elin + (size_t)e * C + f);
  float4 ev = *ep;
  float4 msg;
  msg.x = fmaxf(hv.x + ev.x, 0.0f) + 1e-7f;
  msg.y = fmaxf(hv.y + ev.y, 0.0f) + 1e-7f;
  msg.z = fmaxf(hv.z + ev.z, 0.0f) + 1e-7f;
  msg.w = fmaxf(hv.w + ev.w, 0.0f) + 1e-7f;
  *ep = msg;

  unsigned int* mb = mbits + (size_t)d * C + f;
  atomicMax(mb + 0, __float_as_uint(msg.x));
  atomicMax(mb + 1, __float_as_uint(msg.y));
  atomicMax(mb + 2, __float_as_uint(msg.z));
  atomicMax(mb + 3, __float_as_uint(msg.w));
}

// ---------------------------------------------------------------------------
// Pass B over edges: e = exp(msg - m[dst]); S[dst] += e; T[dst] += msg*e.
// (alpha normalization folded into node kernel: agg = T/(S+1e-16).)
// ---------------------------------------------------------------------------
__global__ void edge_exp_sum_kernel(const float* __restrict__ Msg,
                                    const int* __restrict__ dstI,
                                    const float* __restrict__ Mx,
                                    float* __restrict__ S,
                                    float* __restrict__ T,
                                    int C, long long total) {
  long long tid = (long long)blockIdx.x * blockDim.x + threadIdx.x;
  if (tid >= total) return;
  const int c4 = C >> 2;
  const long long e = tid / c4;
  const int f = (int)(tid % c4) << 2;
  const int d = dstI[e];

  const float4 msg = *(const float4*)(Msg + (size_t)e * C + f);
  const float4 mv  = *(const float4*)(Mx + (size_t)d * C + f);
  float ex = __expf(msg.x - mv.x);
  float ey = __expf(msg.y - mv.y);
  float ez = __expf(msg.z - mv.z);
  float ew = __expf(msg.w - mv.w);

  float* sp = S + (size_t)d * C + f;
  float* tp = T + (size_t)d * C + f;
  atomicAdd(sp + 0, ex); atomicAdd(sp + 1, ey);
  atomicAdd(sp + 2, ez); atomicAdd(sp + 3, ew);
  atomicAdd(tp + 0, msg.x * ex); atomicAdd(tp + 1, msg.y * ey);
  atomicAdd(tp + 2, msg.z * ez); atomicAdd(tp + 3, msg.w * ew);
}

// out[i,f] = T/(S+1e-16) + Hdst[i,f]   (residual with dst features)
__global__ void node_agg_kernel(const float* __restrict__ T,
                                const float* __restrict__ S,
                                const float* __restrict__ Hdst,
                                float* __restrict__ Out,
                                long long total) {
  long long tid = (long long)blockIdx.x * blockDim.x + threadIdx.x;
  if (tid >= total) return;
  Out[tid] = T[tid] / (S[tid] + 1e-16f) + Hdst[tid];
}

// ------------------------- BatchNorm (training mode) -----------------------
__global__ void bn_reduce_kernel(const float* __restrict__ H,
                                 float* __restrict__ sums,
                                 float* __restrict__ sumsq,
                                 int Nrows, int C) {
  const int col = threadIdx.x;
  if (col >= C) return;
  float s = 0.0f, q = 0.0f;
  for (int r = blockIdx.x; r < Nrows; r += gridDim.x) {
    const float v = H[(size_t)r * C + col];
    s += v; q += v * v;
  }
  atomicAdd(&sums[col], s);
  atomicAdd(&sumsq[col], q);
}

__global__ void bn_finalize_kernel(const float* __restrict__ sums,
                                   const float* __restrict__ sumsq,
                                   const float* __restrict__ g,
                                   const float* __restrict__ be,
                                   float* __restrict__ scale,
                                   float* __restrict__ shift,
                                   int Nrows, int C) {
  const int c = threadIdx.x;
  if (c >= C) return;
  const float inv = 1.0f / (float)Nrows;
  const float mu = sums[c] * inv;
  const float var = sumsq[c] * inv - mu * mu;       // biased, matches h.var(0)
  const float sc = g[c] * rsqrtf(var + 1e-5f);
  scale[c] = sc;
  shift[c] = be[c] - mu * sc;
}

__global__ void bn_apply_relu_kernel(float* __restrict__ H,
                                     const float* __restrict__ scale,
                                     const float* __restrict__ shift,
                                     int C, long long total) {
  long long tid = (long long)blockIdx.x * blockDim.x + threadIdx.x;
  if (tid >= total) return;
  const int col = (int)(tid % C);
  H[tid] = fmaxf(H[tid] * scale[col] + shift[col], 0.0f);
}

// ---------------------------- global mean pool -----------------------------
__global__ void pool_accum_kernel(const float* __restrict__ H,
                                  const int* __restrict__ batch,
                                  float* __restrict__ pool,
                                  float* __restrict__ cnt,
                                  int C, long long total) {
  long long tid = (long long)blockIdx.x * blockDim.x + threadIdx.x;
  if (tid >= total) return;
  const int c4 = C >> 2;
  const long long i = tid / c4;
  const int f = (int)(tid % c4) << 2;
  const int b = batch[i];
  const float4 v = *(const float4*)(H + (size_t)i * C + f);
  float* pp = pool + (size_t)b * C + f;
  atomicAdd(pp + 0, v.x); atomicAdd(pp + 1, v.y);
  atomicAdd(pp + 2, v.z); atomicAdd(pp + 3, v.w);
  if (f == 0) atomicAdd(&cnt[b], 1.0f);
}

// head: g = pool/cnt; g1 = g@d1W+d1b; logits = g1@d2W+d2b; log_softmax.
// one 128-thread block per graph (tiny: 64 graphs)
__global__ void head_kernel(const float* __restrict__ pool,
                            const float* __restrict__ cnt,
                            const float* __restrict__ d1W,
                            const float* __restrict__ d1b,
                            const float* __restrict__ d2W,
                            const float* __restrict__ d2b,
                            float* __restrict__ out) {
  __shared__ float g[128];
  __shared__ float g1[64];
  __shared__ float logits[OUT_DIM];
  __shared__ float lse;
  const int gid = blockIdx.x;
  const int t = threadIdx.x;
  const float c = fmaxf(cnt[gid], 1.0f);
  g[t] = pool[(size_t)gid * 128 + t] / c;
  __syncthreads();
  if (t < 64) {
    float acc = d1b[t];
    for (int k = 0; k < 128; ++k) acc += g[k] * d1W[k * 64 + t];
    g1[t] = acc;
  }
  __syncthreads();
  if (t < OUT_DIM) {
    float acc = d2b[t];
    for (int k = 0; k < 64; ++k) acc += g1[k] * d2W[k * OUT_DIM + t];
    logits[t] = acc;
  }
  __syncthreads();
  if (t == 0) {
    float mx = logits[0];
    for (int i = 1; i < OUT_DIM; ++i) mx = fmaxf(mx, logits[i]);
    float s = 0.0f;
    for (int i = 0; i < OUT_DIM; ++i) s += __expf(logits[i] - mx);
    lse = mx + __logf(s);
  }
  __syncthreads();
  if (t < OUT_DIM) out[(size_t)gid * OUT_DIM + t] = logits[t] - lse;
}

// ---------------------------------------------------------------------------
// Host-side orchestration
// ---------------------------------------------------------------------------
struct ConvW {
  const float *Wsrc, *bsrc, *Wdst, *bdst;
  const float *Wedge, *bedge, *W1, *b1, *g1, *be1, *W2, *b2;
};

struct WSLayout {
  float *h1, *h2, *src, *dst, *elin, *m, *s, *t, *out, *p, *bn, *pool, *cnt;
};

template <bool RELU>
static inline void launch_gemm(const float* A, const float* W, const float* bias,
                               float* C, int M, int N, int K, hipStream_t st) {
  dim3 grid((M + 63) / 64, N / 64);
  wmma_gemm_kernel<RELU><<<grid, 128, 0, st>>>(A, W, bias, C, M, N, K);
}

static inline int blocks_for(long long total, int bs) {
  return (int)((total + bs - 1) / bs);
}

static void run_conv(const float* xin, int cin, int cout, const ConvW& p,
                     const float* edge_attr, const int* src, const int* dst,
                     float* layer_out, const WSLayout& B, hipStream_t st) {
  const float* hs;
  const float* hd;
  if (p.Wsrc) {
    launch_gemm<false>(xin, p.Wsrc, p.bsrc, B.src, NN, cout, cin, st);
    launch_gemm<false>(xin, p.Wdst, p.bdst, B.dst, NN, cout, cin, st);
    hs = B.src; hd = B.dst;
  } else {
    hs = xin; hd = xin;               // c_in == c_out: identity transforms
  }

  // edge feature transform: Elin = edge_attr @ Wedge + bedge
  launch_gemm<false>(edge_attr, p.Wedge, p.bedge, B.elin, EE, cout, EDGE_DIM, st);

  const size_t nc_bytes = (size_t)NN * cout * sizeof(float);
  hipMemsetAsync(B.m, 0, nc_bytes, st);   // 0x0 == 0.0f == exact -inf substitute
  hipMemsetAsync(B.s, 0, nc_bytes, st);
  hipMemsetAsync(B.t, 0, nc_bytes, st);

  const long long etot = (long long)EE * (cout >> 2);
  edge_msg_max_kernel<<<blocks_for(etot, 256), 256, 0, st>>>(
      hs, B.elin, src, dst, (unsigned int*)B.m, cout, etot);
  edge_exp_sum_kernel<<<blocks_for(etot, 256), 256, 0, st>>>(
      B.elin, dst, B.m, B.s, B.t, cout, etot);

  const long long ntot = (long long)NN * cout;
  node_agg_kernel<<<blocks_for(ntot, 256), 256, 0, st>>>(B.t, B.s, hd, B.out, ntot);

  // internal MLP: Linear(c,2c) -> BN -> ReLU -> Linear(2c,c); outer ReLU fused
  const int C2 = 2 * cout;
  launch_gemm<false>(B.out, p.W1, p.b1, B.p, NN, C2, cout, st);

  hipMemsetAsync(B.bn, 0, 512 * sizeof(float), st);          // sums + sumsq
  bn_reduce_kernel<<<512, C2, 0, st>>>(B.p, B.bn, B.bn + 256, NN, C2);
  bn_finalize_kernel<<<1, C2, 0, st>>>(B.bn, B.bn + 256, p.g1, p.be1,
                                       B.bn + 512, B.bn + 768, NN, C2);
  const long long ptot = (long long)NN * C2;
  bn_apply_relu_kernel<<<blocks_for(ptot, 256), 256, 0, st>>>(
      B.p, B.bn + 512, B.bn + 768, C2, ptot);

  launch_gemm<true>(B.p, p.W2, p.b2, layer_out, NN, cout, C2, st);
}

extern "C" void kernel_launch(void* const* d_in, const int* in_sizes, int n_in,
                              void* d_out, int out_size, void* d_ws, size_t ws_size,
                              hipStream_t stream) {
  (void)in_sizes; (void)n_in; (void)out_size; (void)ws_size;

  // ---- unpack inputs (setup_inputs dict insertion order, leaves in order) --
  int i = 0;
  const float* x         = (const float*)d_in[i++];  // [50000,128]
  const float* edge_attr = (const float*)d_in[i++];  // [800000,32]
  auto nxt = [&]() { return (const float*)d_in[i++]; };

  ConvW c1{}, c2{}, c3{};
  c1.Wsrc = nxt(); c1.bsrc = nxt(); c1.Wdst = nxt(); c1.bdst = nxt();
  c1.Wedge = nxt(); c1.bedge = nxt(); c1.W1 = nxt(); c1.b1 = nxt();
  c1.g1 = nxt(); c1.be1 = nxt(); c1.W2 = nxt(); c1.b2 = nxt();
  // c2: 64 -> 64, no lin_src/lin_dst
  c2.Wedge = nxt(); c2.bedge = nxt(); c2.W1 = nxt(); c2.b1 = nxt();
  c2.g1 = nxt(); c2.be1 = nxt(); c2.W2 = nxt(); c2.b2 = nxt();
  c3.Wsrc = nxt(); c3.bsrc = nxt(); c3.Wdst = nxt(); c3.bdst = nxt();
  c3.Wedge = nxt(); c3.bedge = nxt(); c3.W1 = nxt(); c3.b1 = nxt();
  c3.g1 = nxt(); c3.be1 = nxt(); c3.W2 = nxt(); c3.b2 = nxt();
  const float* d1W = nxt(); const float* d1b = nxt();
  const float* d2W = nxt(); const float* d2b = nxt();
  const int* edge_index = (const int*)d_in[i++];     // [2, 800000]
  const int* batch      = (const int*)d_in[i++];     // [50000]
  const int* srcI = edge_index;
  const int* dstI = edge_index + EE;

  // ---- workspace layout (floats) ------------------------------------------
  float* ws = (float*)d_ws;
  size_t off = 0;
  auto take = [&](size_t n) { float* p = ws + off; off += n; return p; };
  WSLayout B;
  B.h1   = take((size_t)NN * 128);
  B.h2   = take((size_t)NN * 128);
  B.src  = take((size_t)NN * 128);
  B.dst  = take((size_t)NN * 128);
  B.elin = take((size_t)EE * 128);   // edge messages (in-place over Elin)
  B.m    = take((size_t)NN * 128);
  B.s    = take((size_t)NN * 128);
  B.t    = take((size_t)NN * 128);
  B.out  = take((size_t)NN * 128);
  B.p    = take((size_t)NN * 256);   // MLP hidden
  B.bn   = take(1024);               // sums|sumsq|scale|shift (256 each)
  B.pool = take((size_t)NGRAPH * 128);
  B.cnt  = take(NGRAPH);

  // ---- 3 GENConv layers ----------------------------------------------------
  run_conv(x,    IN_DIM, 64,  c1, edge_attr, srcI, dstI, B.h1, B, stream);
  run_conv(B.h1, 64,     64,  c2, edge_attr, srcI, dstI, B.h2, B, stream);
  run_conv(B.h2, 64,     128, c3, edge_attr, srcI, dstI, B.h1, B, stream);

  // ---- global mean pool + head --------------------------------------------
  hipMemsetAsync(B.pool, 0, (size_t)NGRAPH * 128 * sizeof(float), stream);
  hipMemsetAsync(B.cnt, 0, NGRAPH * sizeof(float), stream);
  const long long ptot = (long long)NN * (128 >> 2);
  pool_accum_kernel<<<blocks_for(ptot, 256), 256, 0, stream>>>(
      B.h1, batch, B.pool, B.cnt, 128, ptot);
  head_kernel<<<NGRAPH, 128, 0, stream>>>(B.pool, B.cnt, d1W, d1b, d2W, d2b,
                                          (float*)d_out);
}
